// transformer_55791625175537
// MI455X (gfx1250) — compile-verified, matrix operand order checked by reading, not run on hardware
//
#include <hip/hip_runtime.h>
#include <math.h>

// Point-transformer layer for MI455X (gfx1250, wave32, WMMA + TDM).
// All dense GEMMs use v_wmma_f32_16x16x32_bf16 (bf16 in, f32 accum).
// Weights are repacked once per call into bf16 wave-tile layout so each
// B-tile is one 32-byte load per lane; A-tiles use 16-byte vector loads.

typedef __attribute__((ext_vector_type(16))) __bf16  v16bf;
typedef __attribute__((ext_vector_type(8)))  float   v8f;
typedef __attribute__((ext_vector_type(4)))  float   v4f;
typedef __attribute__((ext_vector_type(4)))  unsigned int v4u;
typedef __attribute__((ext_vector_type(8)))  int     v8i;
typedef __attribute__((ext_vector_type(4)))  int     v4i;

#define BATCH 4
#define NN    4096
#define CIN   128
#define DIMD  64
#define KNB   16
#define HID   256          // DIM*MULT
#define NPTS  (BATCH*NN)   // 16384
#define NROWS (NPTS*KNB)   // 262144
#define EPS   1e-3f

// ---- WMMA tile helpers (ISA 7.12.2 layouts) --------------------------------
// A (16x32 bf16): lane holds row M=lane&15; K = base + {0..7, 16..23}, base=8*(lane>>4).
// B (32x16 bf16): lane holds col N=lane&15; same K pattern (packed offline).
// C/D (16x16 f32): lane holds col N=lane&15; VGPR r -> row M=r+8*(lane>>4).
__device__ __forceinline__ int kmap(int lane, int e) {
  int i = e >> 1, r = e & 1;
  return ((lane >> 4) << 3) + ((i & 4) << 2) + ((i & 3) << 1) + r;
}

// A-tile: two contiguous 8-float runs per lane -> 4x b128 loads + pk cvt.
__device__ __forceinline__ v16bf load_a16(const float* src, int ld, int lane) {
  int m = lane & 15, base = (lane >> 4) << 3;
  const float* p = src + m * ld + base;
  v4f x0 = *(const v4f*)(p);
  v4f x1 = *(const v4f*)(p + 4);
  v4f x2 = *(const v4f*)(p + 16);
  v4f x3 = *(const v4f*)(p + 20);
  v16bf a;
#pragma unroll
  for (int j = 0; j < 4; ++j) {
    a[j]      = (__bf16)x0[j];
    a[4 + j]  = (__bf16)x1[j];
    a[8 + j]  = (__bf16)x2[j];
    a[12 + j] = (__bf16)x3[j];
  }
  return a;
}

// B-tile from packed bf16 weights: one 32-byte load per lane.
__device__ __forceinline__ v16bf load_b16p(const __bf16* pk, int tile, int lane) {
  return *(const v16bf*)(pk + (((size_t)tile << 5) + lane) * 16);
}

__device__ __forceinline__ v8f wmma_bf16(v16bf a, v16bf b, v8f c) {
  return __builtin_amdgcn_wmma_f32_16x16x32_bf16(false, a, false, b,
                                                 (short)0, c, false, false);
}

// ---- K0: zero stats buffers ------------------------------------------------
__global__ void k_zero(float* st1, float* st2) {
  int t = blockIdx.x * 256 + threadIdx.x;
  if (t < 128) st1[t] = 0.f;
  if (t < 512) st2[t] = 0.f;
}

// ---- K0b: pack fp32 weight (KxN row-major) into bf16 wave-tile layout ------
// out[((kt*ntiles+nt)*32 + lane)*16 + e] = W[(kt*32 + kmap(lane,e))*N + nt*16 + (lane&15)]
__global__ void k_pack(const float* __restrict__ W, int N, int ntiles, int total,
                       __bf16* __restrict__ out) {
  int t = blockIdx.x * 256 + threadIdx.x;
  if (t >= total) return;
  int e = t & 15, lane = (t >> 4) & 31, tile = t >> 9;
  int kt = tile / ntiles, nt = tile - kt * ntiles;
  int kk = kt * 32 + kmap(lane, e);
  int n  = nt * 16 + (lane & 15);
  out[t] = (__bf16)W[kk * N + n];
}

// ---- K1: KNN (top-16 by squared distance, includes self) -------------------
// Candidate cloud (4096x3 f32 = 48KB) staged into LDS via the Tensor Data
// Mover (one 1-D descriptor issued by wave 0, s_wait_tensorcnt, barrier).
__global__ __launch_bounds__(256) void k_knn(const float* __restrict__ pos,
                                             int* __restrict__ idxb) {
  __shared__ float sp[NN * 3];
  int b = blockIdx.y;
  const float* pb = pos + b * NN * 3;

#if __has_builtin(__builtin_amdgcn_tensor_load_to_lds) && \
    __has_builtin(__builtin_amdgcn_s_wait_tensorcnt)
  if (threadIdx.x < 32) {                      // wave 0 issues the TDM op
    unsigned long long ga = (unsigned long long)(const void*)pb;
    unsigned int lds_addr = (unsigned int)(unsigned long long)(void*)sp;
    // D# group0: count=1, lds_addr, 57-bit global addr, type=2 ("image")
    v4u g0 = { 1u, lds_addr, (unsigned int)ga,
               (unsigned int)((ga >> 32) & 0x01FFFFFFu) | (2u << 30) };
    // D# group1: data_size=4B; tensor_dim0=tile_dim0=12288 (1-D copy),
    // tensor_dim1=1, tensor_dim0_stride=12288
    v8i g1 = { (int)(2u << 16),          // data_size=2 (4 bytes)
               (int)(12288u << 16),      // tensor_dim0[15:0] in bits 63:48
               (int)(1u << 16),          // dim0[31:16]=0 | tensor_dim1[15:0]=1
               (int)(12288u << 16),      // tensor_dim1 hi=0 | tile_dim0=12288
               0,                        // tile_dim1=0, tile_dim2=0
               12288,                    // tensor_dim0_stride[31:0]
               0, 0 };
    v4i gz4 = { 0, 0, 0, 0 };
    v8i gz8 = { 0, 0, 0, 0, 0, 0, 0, 0 };
    __builtin_amdgcn_tensor_load_to_lds(g0, g1, gz4, gz4, gz8, 0);
    __builtin_amdgcn_s_wait_tensorcnt(0);
  }
#else
  for (int i = threadIdx.x; i < NN * 3; i += 256) sp[i] = pb[i];
#endif
  __syncthreads();

  int q = blockIdx.x * 256 + threadIdx.x;      // query index within batch
  float qx = sp[q * 3], qy = sp[q * 3 + 1], qz = sp[q * 3 + 2];
  float bd[KNB]; int bi[KNB];
#pragma unroll
  for (int j = 0; j < KNB; ++j) { bd[j] = 3.4e38f; bi[j] = 0; }

  for (int c = 0; c < NN; ++c) {
    float dx = qx - sp[c * 3], dy = qy - sp[c * 3 + 1], dz = qz - sp[c * 3 + 2];
    float d = dx * dx + dy * dy + dz * dz;
    if (d < bd[KNB - 1]) {
#pragma unroll
      for (int j = KNB - 1; j > 0; --j)
        if (d < bd[j - 1]) { bd[j] = bd[j - 1]; bi[j] = bi[j - 1]; }
#pragma unroll
      for (int j = KNB - 1; j >= 0; --j)    // descending: bd[j-1] still pristine
        if (d < bd[j] && (j == 0 || d >= bd[j - 1])) { bd[j] = d; bi[j] = c; }
    }
  }
  int base = (b * NN + q) * KNB;
#pragma unroll
  for (int j = 0; j < KNB; ++j) idxb[base + j] = bi[j];
}

// ---- K2: h = x@W_start+b; key_/value/query projections (WMMA) --------------
__global__ __launch_bounds__(256) void k_start_qkv(
    const float* __restrict__ x,
    const __bf16* __restrict__ pWst, const float* __restrict__ bst,
    const __bf16* __restrict__ pWk,  const float* __restrict__ bk,
    const __bf16* __restrict__ pWq,  const float* __restrict__ bq,
    const __bf16* __restrict__ pWv,  const float* __restrict__ bv,
    float* __restrict__ keyo, float* __restrict__ valo, float* __restrict__ qryo) {
  __shared__ float hbuf[8][16 * DIMD];         // 32 KB, per-wave h tile
  int wave = threadIdx.x >> 5, lane = threadIdx.x & 31;
  int tile = blockIdx.x * 8 + wave;            // 16-row tile of 16384 rows
  int n16 = lane & 15, hi = (lane >> 4) << 3;
  const float* xrow = x + (size_t)tile * 16 * CIN;
  __builtin_prefetch(pWst, 0, 3);              // global_prefetch_b8

  v16bf a[4];
#pragma unroll
  for (int ks = 0; ks < 4; ++ks) a[ks] = load_a16(xrow + ks * 32, CIN, lane);

  float* h = hbuf[wave];
#pragma unroll
  for (int nt = 0; nt < 4; ++nt) {
    v8f c = {};
#pragma unroll
    for (int ks = 0; ks < 4; ++ks)
      c = wmma_bf16(a[ks], load_b16p(pWst, ks * 4 + nt, lane), c);
    int n = nt * 16 + n16; float bb = bst[n];
#pragma unroll
    for (int r = 0; r < 8; ++r) h[(r + hi) * DIMD + n] = c[r] + bb;
  }

  v16bf ha[2];
#pragma unroll
  for (int ks = 0; ks < 2; ++ks) ha[ks] = load_a16(h + ks * 32, DIMD, lane);

  const __bf16* Ws[3] = {pWk, pWq, pWv};
  const float*  Bs[3] = {bk, bq, bv};
  float* Os[3] = {keyo, valo, qryo};  // ref naming: key_=h@Wk, value=h@Wq, query=h@Wv
#pragma unroll
  for (int m3 = 0; m3 < 3; ++m3) {
    float* o = Os[m3] + (size_t)tile * 16 * DIMD;
#pragma unroll
    for (int nt = 0; nt < 4; ++nt) {
      v8f c = {};
#pragma unroll
      for (int ks = 0; ks < 2; ++ks)
        c = wmma_bf16(ha[ks], load_b16p(Ws[m3], ks * 4 + nt, lane), c);
      int n = nt * 16 + n16; float bb = Bs[m3][n];
#pragma unroll
      for (int r = 0; r < 8; ++r) o[(r + hi) * DIMD + n] = c[r] + bb;
    }
  }
}

// ---- K3: BN1 stats over pos_rel@W_pos1+b_pos1 (64 channels) ----------------
__global__ __launch_bounds__(256) void k_stats1(
    const float* __restrict__ pos, const int* __restrict__ idxb,
    const float* __restrict__ W1, const float* __restrict__ b1,
    float* __restrict__ st) {
  __shared__ float red[2][256];
  int c = threadIdx.x & 63, sub = threadIdx.x >> 6;
  float w0 = W1[c], w1 = W1[64 + c], w2 = W1[128 + c], bb = b1[c];
  float s = 0.f, q2 = 0.f;
  for (int row = blockIdx.x * 4 + sub; row < NROWS; row += gridDim.x * 4) {
    int ng = row >> 4;                 // global point index b*N+n
    int b  = ng >> 12;
    int nb = idxb[row];
    const float* p  = pos + ng * 3;
    const float* pc = pos + (b * NN + nb) * 3;
    float dx = p[0] - pc[0], dy = p[1] - pc[1], dz = p[2] - pc[2];
    float t = dx * w0 + dy * w1 + dz * w2 + bb;
    s += t; q2 += t * t;
  }
  red[0][threadIdx.x] = s; red[1][threadIdx.x] = q2;
  __syncthreads();
  if (sub == 0) {
    s  = red[0][c] + red[0][64 + c] + red[0][128 + c] + red[0][192 + c];
    q2 = red[1][c] + red[1][64 + c] + red[1][128 + c] + red[1][192 + c];
    atomicAdd(&st[c], s); atomicAdd(&st[64 + c], q2);
  }
}

__global__ void k_fin1(const float* st, const float* gamma, const float* beta,
                       float* sclsh) {
  int c = threadIdx.x;                 // 64
  float inv = 1.f / (float)NROWS;
  float mean = st[c] * inv;
  float var  = st[64 + c] * inv - mean * mean;
  float scl  = gamma[c] * rsqrtf(var + EPS);
  sclsh[c] = scl;
  sclsh[64 + c] = beta[c] - mean * scl;
}

// ---- K4: BN2 stats: recompute pe, s, t2=s@W_attn1 (WMMA), reduce sum/sumsq -
__global__ __launch_bounds__(128) void k_stats2(
    const float* __restrict__ pos, const int* __restrict__ idxb,
    const float* __restrict__ qry, const float* __restrict__ keyv,
    const float* __restrict__ W1, const float* __restrict__ b1,
    const float* __restrict__ sclsh1,
    const __bf16* __restrict__ pWp2, const float* __restrict__ bp2,
    const __bf16* __restrict__ pWa1, const float* __restrict__ ba1,
    float* __restrict__ st2) {
  __shared__ float bufA[4][16 * DIMD];
  __shared__ float bufB[4][16 * DIMD];
  __shared__ float prel[4][16][3];
  __shared__ int   rowk[4][16];
  __shared__ float lstat[512];
  int wave = threadIdx.x >> 5, lane = threadIdx.x & 31;
  int n16 = lane & 15, hi = (lane >> 4) << 3;
  for (int i = threadIdx.x; i < 512; i += 128) lstat[i] = 0.f;
  __syncthreads();

  int tile = blockIdx.x * 4 + wave;     // == global point index ng
  int row0 = tile * 16;
  if (lane < 16) {
    int row = row0 + lane;
    int b = tile >> 12;
    int nb = idxb[row];
    rowk[wave][lane] = (b * NN + nb) * DIMD;
    const float* p  = pos + tile * 3;
    const float* pc = pos + (b * NN + nb) * 3;
    prel[wave][lane][0] = p[0] - pc[0];
    prel[wave][lane][1] = p[1] - pc[1];
    prel[wave][lane][2] = p[2] - pc[2];
  }
  int qoff = tile * DIMD;
  float* A = bufA[wave]; float* Pe = bufB[wave];

  // t1 -> BN1 -> relu
  for (int t = lane; t < 16 * DIMD; t += 32) {
    int m = t >> 6, c = t & 63;
    float v = prel[wave][m][0] * W1[c] + prel[wave][m][1] * W1[64 + c] +
              prel[wave][m][2] * W1[128 + c] + b1[c];
    v = v * sclsh1[c] + sclsh1[64 + c];
    A[t] = fmaxf(v, 0.f);
  }
  v16bf ta[2];
#pragma unroll
  for (int ks = 0; ks < 2; ++ks) ta[ks] = load_a16(A + ks * 32, DIMD, lane);
#pragma unroll
  for (int nt = 0; nt < 4; ++nt) {      // pe = relu(bn(t1)) @ W_pos2 + b_pos2
    v8f c = {};
#pragma unroll
    for (int ks = 0; ks < 2; ++ks)
      c = wmma_bf16(ta[ks], load_b16p(pWp2, ks * 4 + nt, lane), c);
    int n = nt * 16 + n16; float bb = bp2[n];
#pragma unroll
    for (int r = 0; r < 8; ++r) Pe[(r + hi) * DIMD + n] = c[r] + bb;
  }
  // s = query - key_gathered + pe
  for (int t = lane; t < 16 * DIMD; t += 32) {
    int m = t >> 6, c = t & 63;
    A[t] = qry[qoff + c] - keyv[rowk[wave][m] + c] + Pe[t];
  }
  v16bf sa[2];
#pragma unroll
  for (int ks = 0; ks < 2; ++ks) sa[ks] = load_a16(A + ks * 32, DIMD, lane);
  // t2 = s @ W_attn1 ; accumulate stats of (t2 + b_attn1)
#pragma unroll
  for (int nt = 0; nt < 16; ++nt) {
    v8f c = {};
#pragma unroll
    for (int ks = 0; ks < 2; ++ks)
      c = wmma_bf16(sa[ks], load_b16p(pWa1, ks * 16 + nt, lane), c);
    int ch = nt * 16 + n16;
    float bb = ba1[ch];
    float s = 0.f, q2 = 0.f;
#pragma unroll
    for (int r = 0; r < 8; ++r) { float u = c[r] + bb; s += u; q2 += u * u; }
    s  += __shfl_xor(s, 16);
    q2 += __shfl_xor(q2, 16);
    if (lane < 16) { atomicAdd(&lstat[ch], s); atomicAdd(&lstat[256 + ch], q2); }
  }
  __syncthreads();
  for (int i = threadIdx.x; i < 512; i += 128) atomicAdd(&st2[i], lstat[i]);
}

__global__ void k_fin2(const float* st, const float* gamma, const float* beta,
                       const float* ba1, float* sclsh) {
  int c = threadIdx.x;                  // 256
  float inv = 1.f / (float)NROWS;
  float mean = st[c] * inv;
  float var  = st[256 + c] * inv - mean * mean;
  float scl  = gamma[c] * rsqrtf(var + EPS);
  sclsh[c] = scl;                                   // applied to raw t2 (bias folded)
  sclsh[256 + c] = beta[c] + (ba1[c] - mean) * scl;
}

// ---- K5: full fused attention + aggregation + end GEMM + residual ----------
__global__ __launch_bounds__(128) void k_final(
    const float* __restrict__ x, const float* __restrict__ pos,
    const int* __restrict__ idxb,
    const float* __restrict__ qry, const float* __restrict__ keyv,
    const float* __restrict__ valv,
    const float* __restrict__ W1, const float* __restrict__ b1,
    const float* __restrict__ sclsh1,
    const __bf16* __restrict__ pWp2, const float* __restrict__ bp2,
    const __bf16* __restrict__ pWa1, const float* __restrict__ sclsh2,
    const __bf16* __restrict__ pWa2, const float* __restrict__ ba2,
    const __bf16* __restrict__ pWend, const float* __restrict__ bend,
    float* __restrict__ out) {
  __shared__ float bufA[4][16 * DIMD];
  __shared__ float bufB[4][16 * DIMD];
  __shared__ float bounce[4][16 * 32];
  __shared__ float aggb[16 * DIMD];     // block-level agg rows
  __shared__ float prel[4][16][3];
  __shared__ int   rowk[4][16];
  int wave = threadIdx.x >> 5, lane = threadIdx.x & 31;
  int n16 = lane & 15, hi = (lane >> 4) << 3;
  int ngbase = blockIdx.x * 16;

  for (int sub = 0; sub < 4; ++sub) {   // uniform loop over waves' 4 tiles
    int nsub = wave * 4 + sub;          // n within block (0..15)
    int ng = ngbase + nsub;
    int b = ng >> 12;
    int qoff = ng * DIMD;
    if (lane < 16) {
      int row = ng * KNB + lane;
      int nb = idxb[row];
      rowk[wave][lane] = (b * NN + nb) * DIMD;
      const float* p  = pos + ng * 3;
      const float* pc = pos + (b * NN + nb) * 3;
      prel[wave][lane][0] = p[0] - pc[0];
      prel[wave][lane][1] = p[1] - pc[1];
      prel[wave][lane][2] = p[2] - pc[2];
    }
    float* A = bufA[wave]; float* Pe = bufB[wave]; float* Bo = bounce[wave];

    for (int t = lane; t < 16 * DIMD; t += 32) {
      int m = t >> 6, c = t & 63;
      float v = prel[wave][m][0] * W1[c] + prel[wave][m][1] * W1[64 + c] +
                prel[wave][m][2] * W1[128 + c] + b1[c];
      v = v * sclsh1[c] + sclsh1[64 + c];
      A[t] = fmaxf(v, 0.f);
    }
    v16bf ta[2];
#pragma unroll
    for (int ks = 0; ks < 2; ++ks) ta[ks] = load_a16(A + ks * 32, DIMD, lane);
#pragma unroll
    for (int nt = 0; nt < 4; ++nt) {    // pe
      v8f c = {};
#pragma unroll
      for (int ks = 0; ks < 2; ++ks)
        c = wmma_bf16(ta[ks], load_b16p(pWp2, ks * 4 + nt, lane), c);
      int n = nt * 16 + n16; float bb = bp2[n];
#pragma unroll
      for (int r = 0; r < 8; ++r) Pe[(r + hi) * DIMD + n] = c[r] + bb;
    }
    for (int t = lane; t < 16 * DIMD; t += 32) {  // s = q - k_g + pe
      int m = t >> 6, c = t & 63;
      A[t] = qry[qoff + c] - keyv[rowk[wave][m] + c] + Pe[t];
    }
    v16bf sa[2];
#pragma unroll
    for (int ks = 0; ks < 2; ++ks) sa[ks] = load_a16(A + ks * 32, DIMD, lane);

    // a2 = relu(bn(s@W_attn1)) @ W_attn2, K=256 streamed in 8 chunks of 32
    v8f acc[4] = {};
#pragma unroll
    for (int kc = 0; kc < 8; ++kc) {
#pragma unroll
      for (int s2 = 0; s2 < 2; ++s2) {
        v8f c = {};
#pragma unroll
        for (int ks = 0; ks < 2; ++ks)
          c = wmma_bf16(sa[ks], load_b16p(pWa1, ks * 16 + kc * 2 + s2, lane), c);
        int ch = kc * 32 + s2 * 16 + n16;
        float scl = sclsh2[ch], sh = sclsh2[256 + ch];
#pragma unroll
        for (int r = 0; r < 8; ++r)
          Bo[(r + hi) * 32 + s2 * 16 + n16] = fmaxf(c[r] * scl + sh, 0.f);
      }
      v16bf aa = load_a16(Bo, 32, lane);  // C-layout -> A-layout via LDS bounce
#pragma unroll
      for (int nt2 = 0; nt2 < 4; ++nt2)
        acc[nt2] = wmma_bf16(aa, load_b16p(pWa2, kc * 4 + nt2, lane), acc[nt2]);
    }
#pragma unroll
    for (int nt2 = 0; nt2 < 4; ++nt2) { // a2 + b_attn2 into A (s no longer needed)
      int n = nt2 * 16 + n16; float bb = ba2[n];
#pragma unroll
      for (int r = 0; r < 8; ++r) A[(r + hi) * DIMD + n] = acc[nt2][r] + bb;
    }
    // softmax over 64 channels per (n,k) row
    if (lane < 16) {
      float mx = -3.4e38f;
      for (int c = 0; c < DIMD; ++c) mx = fmaxf(mx, A[lane * DIMD + c]);
      float sm = 0.f;
      for (int c = 0; c < DIMD; ++c) {
        float e = __expf(A[lane * DIMD + c] - mx);
        A[lane * DIMD + c] = e; sm += e;
      }
      float inv = 1.f / sm;
      for (int c = 0; c < DIMD; ++c) A[lane * DIMD + c] *= inv;
    }
    // agg[n][c] = sum_k a[k][c] * (value[n][c] + pe[k][c])
    for (int c = lane; c < DIMD; c += 32) {
      float vv = valv[qoff + c];
      float s = 0.f;
#pragma unroll
      for (int m = 0; m < 16; ++m) s += A[m * DIMD + c] * (vv + Pe[m * DIMD + c]);
      aggb[nsub * DIMD + c] = s;
    }
  }
  __syncthreads();

  // y = agg @ W_end + b_end + x ; 8 N-tiles split across 4 waves
  v16bf ea[2];
#pragma unroll
  for (int ks = 0; ks < 2; ++ks) ea[ks] = load_a16(aggb + ks * 32, DIMD, lane);
#pragma unroll
  for (int t = 0; t < 2; ++t) {
    int nt = wave * 2 + t;
    v8f c = {};
#pragma unroll
    for (int ks = 0; ks < 2; ++ks)
      c = wmma_bf16(ea[ks], load_b16p(pWend, ks * 8 + nt, lane), c);
    int n = nt * 16 + n16; float bb = bend[n];
#pragma unroll
    for (int r = 0; r < 8; ++r) {
      int m = r + hi;
      size_t o = (size_t)(ngbase + m) * CIN + n;
      out[o] = c[r] + bb + x[o];
    }
  }
}

// ---- host-side launch ------------------------------------------------------
extern "C" void kernel_launch(void* const* d_in, const int* in_sizes, int n_in,
                              void* d_out, int out_size, void* d_ws, size_t ws_size,
                              hipStream_t stream) {
  (void)in_sizes; (void)n_in; (void)out_size; (void)ws_size;
  const float* x    = (const float*)d_in[0];
  const float* pos  = (const float*)d_in[1];
  const float* Wst  = (const float*)d_in[2];
  const float* bst  = (const float*)d_in[3];
  const float* Wk   = (const float*)d_in[4];
  const float* bk   = (const float*)d_in[5];
  const float* Wq   = (const float*)d_in[6];
  const float* bq   = (const float*)d_in[7];
  const float* Wv   = (const float*)d_in[8];
  const float* bv   = (const float*)d_in[9];
  const float* Wp1  = (const float*)d_in[10];
  const float* bp1  = (const float*)d_in[11];
  const float* gp   = (const float*)d_in[12];
  const float* bep  = (const float*)d_in[13];
  const float* Wp2  = (const float*)d_in[14];
  const float* bp2  = (const float*)d_in[15];
  const float* Wa1  = (const float*)d_in[16];
  const float* ba1  = (const float*)d_in[17];
  const float* ga   = (const float*)d_in[18];
  const float* bea  = (const float*)d_in[19];
  const float* Wa2  = (const float*)d_in[20];
  const float* ba2  = (const float*)d_in[21];
  const float* Wend = (const float*)d_in[22];
  const float* bend = (const float*)d_in[23];
  float* out = (float*)d_out;

  // workspace carve (~13.2 MB)
  char* ws = (char*)d_ws;
  int*   idxb   = (int*)ws;                         // NROWS ints
  float* keyo   = (float*)(ws + (size_t)NROWS * sizeof(int));
  float* valo   = keyo + (size_t)NPTS * DIMD;
  float* qryo   = valo + (size_t)NPTS * DIMD;
  float* st1    = qryo + (size_t)NPTS * DIMD;       // 128
  float* st2    = st1 + 128;                        // 512
  float* sclsh1 = st2 + 512;                        // 128
  float* sclsh2 = sclsh1 + 128;                     // 512
  __bf16* pWst  = (__bf16*)(sclsh2 + 512);          // packed bf16 weights
  __bf16* pWk   = pWst + 128 * 64;
  __bf16* pWq   = pWk  + 64 * 64;
  __bf16* pWv   = pWq  + 64 * 64;
  __bf16* pWp2  = pWv  + 64 * 64;
  __bf16* pWa1  = pWp2 + 64 * 64;
  __bf16* pWa2  = pWa1 + 64 * 256;
  __bf16* pWend = pWa2 + 256 * 64;

  k_zero<<<2, 256, 0, stream>>>(st1, st2);
  // pack all weights into bf16 wave-tile layout (L2-resident, 128 KB total)
  k_pack<<<32, 256, 0, stream>>>(Wst,  64,  4,  128 * 64, pWst);
  k_pack<<<16, 256, 0, stream>>>(Wk,   64,  4,  64 * 64,  pWk);
  k_pack<<<16, 256, 0, stream>>>(Wq,   64,  4,  64 * 64,  pWq);
  k_pack<<<16, 256, 0, stream>>>(Wv,   64,  4,  64 * 64,  pWv);
  k_pack<<<16, 256, 0, stream>>>(Wp2,  64,  4,  64 * 64,  pWp2);
  k_pack<<<64, 256, 0, stream>>>(Wa1,  256, 16, 64 * 256, pWa1);
  k_pack<<<64, 256, 0, stream>>>(Wa2,  64,  4,  256 * 64, pWa2);
  k_pack<<<32, 256, 0, stream>>>(Wend, 128, 8,  64 * 128, pWend);

  k_knn<<<dim3(16, BATCH), 256, 0, stream>>>(pos, idxb);
  k_start_qkv<<<NPTS / 16 / 8, 256, 0, stream>>>(x, pWst, bst, pWk, bk, pWq, bq,
                                                 pWv, bv, keyo, valo, qryo);
  k_stats1<<<2048, 256, 0, stream>>>(pos, idxb, Wp1, bp1, st1);
  k_fin1<<<1, 64, 0, stream>>>(st1, gp, bep, sclsh1);
  k_stats2<<<NROWS / 16 / 4, 128, 0, stream>>>(pos, idxb, qryo, keyo, Wp1, bp1,
                                               sclsh1, pWp2, bp2, pWa1, ba1, st2);
  k_fin2<<<1, 256, 0, stream>>>(st2, ga, bea, ba1, sclsh2);
  k_final<<<NPTS / 16, 128, 0, stream>>>(x, pos, idxb, qryo, keyo, valo, Wp1,
                                         bp1, sclsh1, pWp2, bp2, pWa1, sclsh2,
                                         pWa2, ba2, pWend, bend, out);
}
